// ImprovedRecovery_46359876993728
// MI455X (gfx1250) — compile-verified, matrix operand order checked by reading, not run on hardware
//
#include <hip/hip_runtime.h>
#include <hip/hip_bf16.h>
#include <math.h>

typedef __attribute__((ext_vector_type(16))) _Float16 v16h;
typedef __attribute__((ext_vector_type(8)))  _Float16 v8h;
typedef __attribute__((ext_vector_type(8)))  float    v8f;

#define HID   128
#define THREE 384         // 3*HID
#define TLEN  2048
#define NT    24          // N tiles of 16 in 3H
#define KT    4           // K tiles of 32 in 128
#define FRAG_HALVES (NT * KT * 32 * 16)   // 49152 halves = 96KB per matrix

// ---------------------------------------------------------------------------
// Fragment layouts (CDNA5 16-bit WMMA, wave32):
//   A (16x32, MxK): lane L -> M = L&15 ;  half j -> K = (L>>4)*8 + (j<8 ? j : j+8)
//   B (32x16, KxN): lane L -> N = L&15 ;  half j -> K = (L>>4)*8 + (j<8 ? j : j+8)
//   C (16x16 f32) : vgpr i, lane L -> M = i + 8*(L>>4), N = L&15
// Packed B fragment storage: [ntile][ktile][lane][16 halves]  (32B per lane).
// ---------------------------------------------------------------------------

__device__ __forceinline__ void async_to_lds_b128(unsigned lds_byte_off,
                                                  const void* gaddr) {
  // GLOBAL_LOAD_ASYNC_TO_LDS_B128 (GV mode): vdst = LDS byte offset VGPR,
  // vaddr = 64-bit global address. Tracked with ASYNCcnt.
  asm volatile("global_load_async_to_lds_b128 %0, %1, off"
               :: "v"(lds_byte_off), "v"((unsigned long long)gaddr)
               : "memory");
}

__device__ __forceinline__ void wait_async0() {
#if __has_builtin(__builtin_amdgcn_s_wait_asynccnt)
  __builtin_amdgcn_s_wait_asynccnt(0);
#else
  asm volatile("s_wait_asynccnt 0x0" ::: "memory");
#endif
}

// Pre-swizzle one f32 weight matrix [128][384] into f16 B fragments.
__global__ __launch_bounds__(256) void pack_weights(const float* __restrict__ W,
                                                    _Float16* __restrict__ out) {
  int idx  = blockIdx.x * blockDim.x + threadIdx.x;   // 49152 elements
  if (idx >= FRAG_HALVES) return;
  int j    = idx & 15;
  int lane = (idx >> 4) & 31;
  int kt   = (idx >> 9) & 3;
  int nt   = idx >> 11;
  int N    = nt * 16 + (lane & 15);
  int K    = kt * 32 + ((lane >> 4) << 3) + (j < 8 ? j : j + 8);
  out[idx] = (_Float16)W[K * THREE + N];
}

// xw = xin @ W + b_in.  xin:[BT,128] f32, pk: packed f16 frags, xw:[BT,384] f32.
// Packed B (96KB) is staged block-wide into LDS via async-to-LDS; inner loop
// is pure ds_load_b128 -> WMMA -> stores (bias pre-hoisted into registers).
__global__ __launch_bounds__(256) void gru_proj(const float* __restrict__ xin,
                                                const _Float16* __restrict__ pk,
                                                const float* __restrict__ bin,
                                                float* __restrict__ xw) {
  extern __shared__ _Float16 bsh[];                 // FRAG_HALVES (96KB dynamic)
  const int tid  = threadIdx.x;
  const int lane = tid & 31;
  const int wave = tid >> 5;
  const int mt   = blockIdx.x * 8 + wave;
  const int Mrow = lane & 15;
  const int khi  = (lane >> 4) * 8;

  // Stage packed B: 6144 x 16B chunks, 24 per thread, ASYNCcnt tracked.
  {
    const char* src = (const char*)pk;
#pragma unroll
    for (int c = 0; c < 24; ++c) {
      const unsigned chunk = (unsigned)(c * 256 + tid);
      async_to_lds_b128(chunk * 16u, src + (size_t)chunk * 16);
    }
  }

  // Loop-invariant per-lane bias values (one per N tile).
  float bvals[NT];
#pragma unroll
  for (int n = 0; n < NT; ++n) bvals[n] = bin[n * 16 + (lane & 15)];

  // Load + convert the 4 A fragments while the async DMA runs.
  const float* arow = xin + (size_t)(mt * 16 + Mrow) * HID;
  v16h a[KT];
#pragma unroll
  for (int kf = 0; kf < KT; ++kf) {
    const int base = kf * 32 + khi;
#pragma unroll
    for (int j = 0; j < 8; ++j) {
      a[kf][j]     = (_Float16)arow[base + j];
      a[kf][j + 8] = (_Float16)arow[base + 16 + j];
    }
  }

  wait_async0();
  __syncthreads();

  const int cM = 8 * (lane >> 4);
#pragma unroll 1
  for (int n = 0; n < NT; ++n) {
    // Issue all 8 ds_load_b128 up front so the compiler can use progressive
    // s_wait_dscnt waits and overlap LDS latency with earlier WMMAs.
    v16h b[KT];
#pragma unroll
    for (int kf = 0; kf < KT; ++kf)
      b[kf] = *(const v16h*)&bsh[(size_t)((n * KT + kf) * 32 + lane) * 16];

    const float bv = bvals[n];
    v8f c = {bv, bv, bv, bv, bv, bv, bv, bv};      // fold input bias into C
#pragma unroll
    for (int kf = 0; kf < KT; ++kf)
      c = __builtin_amdgcn_wmma_f32_16x16x32_f16(false, a[kf], false, b[kf],
                                                 (short)0, c, false, false);
#pragma unroll
    for (int i = 0; i < 8; ++i)
      xw[(size_t)(mt * 16 + i + cM) * THREE + n * 16 + (lane & 15)] = c[i];
  }
}

// Sequential GRU scan. One block per 16 batch rows (grid = B/16 = 4 blocks).
// Wave w owns H columns 16w..16w+15 for gates z, r, h~ (N tiles w, 8+w, 16+w),
// so gate math is wave-local and h_prev lives in C-layout registers.
// xw is register double-buffered: loads for t+1 issue before the t barriers.
__global__ __launch_bounds__(256) void gru_scan(const float* __restrict__ xw,
                                                const _Float16* __restrict__ prk,
                                                const float* __restrict__ brec,
                                                float* __restrict__ hout) {
  __shared__ _Float16 hsh[16 * HID];                // h_t in f16 (A operand)

  const int tid  = threadIdx.x;
  const int lane = tid & 31;
  const int wave = tid >> 5;
  const int bt   = blockIdx.x;
  const int Ncol = lane & 15;
  const int hi   = lane >> 4;
  const int cM   = 8 * hi;

  // Recurrent weight fragments pinned in registers for the whole scan.
  v16h bfrag[3][KT];
#pragma unroll
  for (int g = 0; g < 3; ++g) {
    const int nt = g * 8 + wave;
#pragma unroll
    for (int kf = 0; kf < KT; ++kf)
      bfrag[g][kf] = *(const v16h*)(prk + (size_t)((nt * KT + kf) * 32 + lane) * 16);
  }
  const float brz = brec[          wave * 16 + Ncol];
  const float brr = brec[HID     + wave * 16 + Ncol];
  const float brh = brec[2 * HID + wave * 16 + Ncol];

  v8f hprev = {};
  for (int k = tid; k < 16 * HID; k += 256) hsh[k] = (_Float16)0.f;
  __syncthreads();

  // Per-element xw row bases (batch row varies with accumulator slot i).
  size_t rbase[8];
#pragma unroll
  for (int i = 0; i < 8; ++i)
    rbase[i] = (size_t)(bt * 16 + i + cM) * TLEN * THREE + wave * 16 + Ncol;

  // Prime the xw double buffer with t = 0.
  float xz[8], xr[8], xh[8];
#pragma unroll
  for (int i = 0; i < 8; ++i) {
    xz[i] = xw[rbase[i]];
    xr[i] = xw[rbase[i] + HID];
    xh[i] = xw[rbase[i] + 2 * HID];
  }

#pragma unroll 1
  for (int t = 0; t < TLEN; ++t) {
    // A fragments (h_t) from LDS: two 16B runs per fragment -> ds_load_b128.
    v16h a[KT];
#pragma unroll
    for (int kf = 0; kf < KT; ++kf) {
      const int base = kf * 32 + hi * 8;
      v8h lo = *(const v8h*)&hsh[(lane & 15) * HID + base];
      v8h hv = *(const v8h*)&hsh[(lane & 15) * HID + base + 16];
#pragma unroll
      for (int j = 0; j < 8; ++j) { a[kf][j] = lo[j]; a[kf][j + 8] = hv[j]; }
    }

    // Three independent accumulator chains interleave in the XDL pipe.
    v8f cz = {}, cr = {}, ch = {};
#pragma unroll
    for (int kf = 0; kf < KT; ++kf) {
      cz = __builtin_amdgcn_wmma_f32_16x16x32_f16(false, a[kf], false, bfrag[0][kf],
                                                  (short)0, cz, false, false);
      cr = __builtin_amdgcn_wmma_f32_16x16x32_f16(false, a[kf], false, bfrag[1][kf],
                                                  (short)0, cr, false, false);
      ch = __builtin_amdgcn_wmma_f32_16x16x32_f16(false, a[kf], false, bfrag[2][kf],
                                                  (short)0, ch, false, false);
    }

    // Issue next step's xw loads now; their latency hides behind gate math,
    // both barriers and the next step's LDS traffic.
    float nz[8], nr[8], nh[8];
    if (t + 1 < TLEN) {
#pragma unroll
      for (int i = 0; i < 8; ++i) {
        const size_t off = rbase[i] + (size_t)(t + 1) * THREE;
        nz[i] = xw[off];
        nr[i] = xw[off + HID];
        nh[i] = xw[off + 2 * HID];
      }
    }

    // Gate math (reset_after=True): h~ = tanh(xh + r*(h@Wh + brh)).
    v8f hnew;
#pragma unroll
    for (int i = 0; i < 8; ++i) {
      const float z  = 1.f / (1.f + __expf(-(xz[i] + cz[i] + brz)));
      const float r  = 1.f / (1.f + __expf(-(xr[i] + cr[i] + brr)));
      const float hh = tanhf(xh[i] + r * (ch[i] + brh));
      hnew[i] = z * hprev[i] + (1.f - z) * hh;
    }
    hprev = hnew;

    __syncthreads();                                // all reads of old h done
#pragma unroll
    for (int i = 0; i < 8; ++i) {
      hsh[(i + cM) * HID + wave * 16 + Ncol] = (_Float16)hnew[i];
      hout[((size_t)(bt * 16 + i + cM) * TLEN + t) * HID + wave * 16 + Ncol] = hnew[i];
    }
    __syncthreads();                                // h_{t+1} visible

    if (t + 1 < TLEN) {
#pragma unroll
      for (int i = 0; i < 8; ++i) { xz[i] = nz[i]; xr[i] = nr[i]; xh[i] = nh[i]; }
    }
  }
}

// out = h @ wo + bo  (128 -> 5), memory bound.
__global__ __launch_bounds__(256) void out_proj(const float* __restrict__ h,
                                                const float* __restrict__ wo,
                                                const float* __restrict__ bo,
                                                float* __restrict__ out, int n) {
  int r = blockIdx.x * blockDim.x + threadIdx.x;
  if (r >= n) return;
  float acc[5];
#pragma unroll
  for (int o = 0; o < 5; ++o) acc[o] = bo[o];
  const float* hr = h + (size_t)r * HID;
#pragma unroll 4
  for (int k = 0; k < HID; ++k) {
    const float hv = hr[k];
#pragma unroll
    for (int o = 0; o < 5; ++o) acc[o] += hv * wo[k * 5 + o];
  }
#pragma unroll
  for (int o = 0; o < 5; ++o) out[(size_t)r * 5 + o] = acc[o];
}

extern "C" void kernel_launch(void* const* d_in, const int* in_sizes, int n_in,
                              void* d_out, int out_size, void* d_ws, size_t ws_size,
                              hipStream_t stream) {
  const float* x  = (const float*)d_in[0];
  const float* k[3]  = {(const float*)d_in[1], (const float*)d_in[4], (const float*)d_in[7]};
  const float* rk[3] = {(const float*)d_in[2], (const float*)d_in[5], (const float*)d_in[8]};
  const float* b[3]  = {(const float*)d_in[3], (const float*)d_in[6], (const float*)d_in[9]};
  const float* wo = (const float*)d_in[10];
  const float* bo = (const float*)d_in[11];
  float* out = (float*)d_out;

  const int BT = in_sizes[0] / HID;            // B*T = 131072
  const int B  = BT / TLEN;                    // 64

  // Workspace layout (bytes):
  //   [0]            packed weights: 6 matrices x 96KB (k0,rk0,k1,rk1,k2,rk2)
  //   [1MB]          xw buffer   : BT*384 f32  (201 MB)
  //   [1MB+201MB]    hA          : BT*128 f32  (64 MB)
  //   [.. + 64MB]    hB          : BT*128 f32  (64 MB)
  _Float16* pw  = (_Float16*)d_ws;
  float* xwbuf  = (float*)((char*)d_ws + (1u << 20));
  float* hA     = (float*)((char*)xwbuf + (size_t)BT * THREE * sizeof(float));
  float* hB     = (float*)((char*)hA    + (size_t)BT * HID   * sizeof(float));

  // 1) Pre-swizzle all six weight matrices into f16 WMMA B fragments.
  const float* srcw[6] = {k[0], rk[0], k[1], rk[1], k[2], rk[2]};
  for (int m = 0; m < 6; ++m)
    pack_weights<<<(FRAG_HALVES + 255) / 256, 256, 0, stream>>>(
        srcw[m], pw + (size_t)m * FRAG_HALVES);

  // 2) Three GRU layers: parallel input GEMM (async-LDS staged weights),
  //    then latency-optimized scan.
  const size_t bshBytes = (size_t)FRAG_HALVES * sizeof(_Float16);   // 96KB
  const float* layer_in = x;
  float* layer_out[3] = {hA, hB, hA};
  for (int l = 0; l < 3; ++l) {
    const _Float16* pk_l  = pw + (size_t)(2 * l)     * FRAG_HALVES;
    const _Float16* prk_l = pw + (size_t)(2 * l + 1) * FRAG_HALVES;
    gru_proj<<<BT / (16 * 8), 256, bshBytes, stream>>>(layer_in, pk_l, b[l], xwbuf);
    gru_scan<<<B / 16, 256, 0, stream>>>(xwbuf, prk_l, b[l] + THREE, layer_out[l]);
    layer_in = layer_out[l];
  }

  // 3) Final 128 -> 5 projection.
  out_proj<<<(BT + 255) / 256, 256, 0, stream>>>(layer_out[2], wo, bo, out, BT);
}
